// StructInductionNoSplit_5669356834056
// MI455X (gfx1250) — compile-verified, matrix operand order checked by reading, not run on hardware
//
#include <hip/hip_runtime.h>
#include <hip/hip_bf16.h>
#include <stdint.h>

#define DI __device__ __forceinline__

typedef __attribute__((ext_vector_type(16))) __bf16       v16bf;
typedef __attribute__((ext_vector_type(8)))  float        v8f;
typedef __attribute__((ext_vector_type(4)))  unsigned int u32x4;

union FragU { u32x4 u[2]; v16bf v; };

static constexpr int    Bn = 48, Tn = 256, Dn = 1024;
static constexpr int    BT = Bn * Tn;                       // 12288
static constexpr size_t OFF_XB  = 0;                        // bf16 X       25,165,824 B
static constexpr size_t OFF_WKB = 25165824;                 // bf16 Wk       2,097,152 B
static constexpr size_t OFF_WQB = 27262976;                 // bf16 Wq       2,097,152 B
static constexpr size_t OFF_KB  = 29360128;                 // bf16 key     25,165,824 B
static constexpr size_t OFF_QB  = 54525952;                 // bf16 query   25,165,824 B
static constexpr size_t OFF_FI  = 79691776;                 // f32 f_i          49,152 B
static constexpr size_t OFF_AG  = 79740928;                 // f32 A        12,582,912 B
// total workspace: 92,323,840 B

// ---------------- fp32 -> bf16 conversion (X, Wk, Wq) ----------------
__global__ void __launch_bounds__(256)
cvt_bf16_kernel(const float* __restrict__ src, __bf16* __restrict__ dst, int n)
{
    int i = (blockIdx.x * 256 + threadIdx.x) * 4;
    if (i + 3 < n) {
        float4 v = *(const float4*)(src + i);
        dst[i + 0] = (__bf16)v.x;
        dst[i + 1] = (__bf16)v.y;
        dst[i + 2] = (__bf16)v.z;
        dst[i + 3] = (__bf16)v.w;
    }
}

// ---------------- f_i = X @ Wr^T + br  (GEMV, one wave32 per row) ----------------
__global__ void __launch_bounds__(256)
fi_kernel(const float* __restrict__ X, const float* __restrict__ Wr,
          const float* __restrict__ br, float* __restrict__ fi)
{
    const int row  = blockIdx.x * 8 + (threadIdx.x >> 5);
    const int lane = threadIdx.x & 31;
    const float* xr = X + (size_t)row * Dn;
    float s = 0.f;
    #pragma unroll 8
    for (int c = 0; c < 32; ++c) {
        int d = lane + 32 * c;
        s += xr[d] * Wr[d];
    }
    #pragma unroll
    for (int off = 16; off > 0; off >>= 1) s += __shfl_xor(s, off, 32);
    if (lane == 0) fi[row] = s + br[0];
}

// ---------------- bf16 A-fragment load per ISA 16x32 layout ----------------
DI v16bf load_frag16x32(const __bf16* rowptr, int half)
{
    FragU f;
    const __bf16* p = rowptr + (half ? 8 : 0);
    f.u[0] = *(const u32x4*)(p);
    f.u[1] = *(const u32x4*)(p + 16);
    return f.v;
}

// ---------------- C = X @ W^T + bias, scaled, stored bf16 (key / query) ----------------
// Each wave computes a 32(M) x 64(N) macro-tile: 2x4 WMMA 16x16 tiles, K-loop of 32.
// All six fragments of a k-step live in distinct registers so the scheduler can use
// staggered s_wait_loadcnt values instead of draining to zero before each WMMA.
__global__ void __launch_bounds__(256)
gemm_kq_kernel(const __bf16* __restrict__ X, const __bf16* __restrict__ W,
               const float* __restrict__ bias, float scale,
               __bf16* __restrict__ Out)
{
    const int wid  = blockIdx.x * 8 + (threadIdx.x >> 5);
    const int lane = threadIdx.x & 31;
    const int half = lane >> 4;
    const int r    = lane & 15;
    const int ng   = wid & 15;        // 16 column groups of 64
    const int mg   = wid >> 4;        // 384 row groups of 32
    const int m0   = mg * 32, n0 = ng * 64;

    v8f acc[2][4];
    #pragma unroll
    for (int nf = 0; nf < 4; ++nf) {
        const float bv = bias[n0 + nf * 16 + r];
        #pragma unroll
        for (int mf = 0; mf < 2; ++mf)
            #pragma unroll
            for (int e = 0; e < 8; ++e) acc[mf][nf][e] = bv;
    }

    const __bf16* x0 = X + (size_t)(m0 + r) * Dn;
    const __bf16* x1 = X + (size_t)(m0 + 16 + r) * Dn;
    const __bf16* w0 = W + (size_t)(n0 + r) * Dn;

    for (int kb = 0; kb < Dn; kb += 32) {
        v16bf a0 = load_frag16x32(x0 + kb, half);
        v16bf a1 = load_frag16x32(x1 + kb, half);
        v16bf bf[4];
        #pragma unroll
        for (int nf = 0; nf < 4; ++nf)
            bf[nf] = load_frag16x32(w0 + (size_t)(nf * 16) * Dn + kb, half);
        __builtin_prefetch(x0 + kb + 64, 0, 3);   // global_prefetch_b8, near scope
        __builtin_prefetch(x1 + kb + 64, 0, 3);
        #pragma unroll
        for (int nf = 0; nf < 4; ++nf) {
            acc[0][nf] = __builtin_amdgcn_wmma_f32_16x16x32_bf16(
                false, a0, false, bf[nf], (short)0, acc[0][nf], false, false);
            acc[1][nf] = __builtin_amdgcn_wmma_f32_16x16x32_bf16(
                false, a1, false, bf[nf], (short)0, acc[1][nf], false, false);
        }
    }

    #pragma unroll
    for (int mf = 0; mf < 2; ++mf)
        #pragma unroll
        for (int nf = 0; nf < 4; ++nf)
            #pragma unroll
            for (int e = 0; e < 8; ++e) {
                int row = m0 + mf * 16 + e + half * 8;
                int col = n0 + nf * 16 + r;
                Out[(size_t)row * Dn + col] = (__bf16)(acc[mf][nf][e] * scale);
            }
}

// ---------------- A = exp(Q @ K^T), zero diagonal, fp32 out ----------------
// Per batch: 256x256, each wave a 32x32 macro-tile (2x2 WMMA), K-loop 1024.
__global__ void __launch_bounds__(256)
gemm_attn_kernel(const __bf16* __restrict__ Q, const __bf16* __restrict__ K,
                 float* __restrict__ A)
{
    const int wid  = blockIdx.x * 8 + (threadIdx.x >> 5);
    const int lane = threadIdx.x & 31;
    const int half = lane >> 4;
    const int r    = lane & 15;
    const int b    = wid >> 6;        // 64 wave-jobs per batch
    const int tj   = wid & 63;
    const int t0   = (tj >> 3) * 32, s0 = (tj & 7) * 32;

    const __bf16* Qb = Q + (size_t)b * Tn * Dn;
    const __bf16* Kb = K + (size_t)b * Tn * Dn;

    v8f acc[2][2];
    #pragma unroll
    for (int mf = 0; mf < 2; ++mf)
        #pragma unroll
        for (int nf = 0; nf < 2; ++nf)
            #pragma unroll
            for (int e = 0; e < 8; ++e) acc[mf][nf][e] = 0.f;

    const __bf16* q0 = Qb + (size_t)(t0 + r) * Dn;
    const __bf16* q1 = Qb + (size_t)(t0 + 16 + r) * Dn;
    const __bf16* k0 = Kb + (size_t)(s0 + r) * Dn;
    const __bf16* k1 = Kb + (size_t)(s0 + 16 + r) * Dn;

    for (int kb = 0; kb < Dn; kb += 32) {
        v16bf a0 = load_frag16x32(q0 + kb, half);
        v16bf a1 = load_frag16x32(q1 + kb, half);
        v16bf b0 = load_frag16x32(k0 + kb, half);
        v16bf b1 = load_frag16x32(k1 + kb, half);
        __builtin_prefetch(q0 + kb + 64, 0, 3);
        __builtin_prefetch(k0 + kb + 64, 0, 3);
        acc[0][0] = __builtin_amdgcn_wmma_f32_16x16x32_bf16(false, a0, false, b0, (short)0, acc[0][0], false, false);
        acc[1][0] = __builtin_amdgcn_wmma_f32_16x16x32_bf16(false, a1, false, b0, (short)0, acc[1][0], false, false);
        acc[0][1] = __builtin_amdgcn_wmma_f32_16x16x32_bf16(false, a0, false, b1, (short)0, acc[0][1], false, false);
        acc[1][1] = __builtin_amdgcn_wmma_f32_16x16x32_bf16(false, a1, false, b1, (short)0, acc[1][1], false, false);
    }

    float* Ab = A + (size_t)b * Tn * Tn;
    #pragma unroll
    for (int mf = 0; mf < 2; ++mf)
        #pragma unroll
        for (int nf = 0; nf < 2; ++nf)
            #pragma unroll
            for (int e = 0; e < 8; ++e) {
                int t = t0 + mf * 16 + e + half * 8;
                int s = s0 + nf * 16 + r;
                float v = (t == s) ? 0.f : __expf(acc[mf][nf][e]);
                Ab[(size_t)t * Tn + s] = v;
            }
}

// ---------------- Laplacian build + in-place Gauss-Jordan inverse + df assembly ----------------
// One 1024-thread workgroup per batch; the full 256x256 fp32 matrix (256 KB) lives in
// CDNA5's 320 KB LDS. The global->LDS staging uses the CDNA5 async-to-LDS path
// (global_load_async_to_lds_b128 + s_wait_asynccnt) so 256 KB moves without VGPR round-trips.
__global__ void __launch_bounds__(1024)
tree_kernel(const float* __restrict__ Ag, const float* __restrict__ fi,
            float* __restrict__ dfo)
{
    extern __shared__ float smem[];
    float* Ls   = smem;             // 65536 floats: L, then L^-1 in place
    float* fcol = smem + 65536;     // 256 floats: colsum / saved pivot column
    float* piv  = smem + 65536 + 256;

    const int N   = 256;
    const int b   = blockIdx.x;
    const int tid = threadIdx.x;
    const int wave = tid >> 5, lane = tid & 31;
    const float* Ab = Ag + (size_t)b * N * N;

    // ---- async copy A[b] (256 KB) global -> LDS: 16384 x b128, 16 issues/thread ----
    {
        // low 32 bits of a generic LDS pointer = wave-relative LDS byte offset
        const uint32_t lds_base = (uint32_t)(uintptr_t)(void*)Ls;
        #pragma unroll
        for (int it = 0; it < 16; ++it) {
            int chunk = tid + it * 1024;                    // 16-byte chunk index
            uint32_t laddr = lds_base + (uint32_t)chunk * 16u;
            uint64_t gaddr = (uint64_t)(uintptr_t)(const void*)(Ab + (size_t)chunk * 4);
            asm volatile("global_load_async_to_lds_b128 %0, %1, off"
                         :: "v"(laddr), "v"(gaddr) : "memory");
        }
        asm volatile("s_wait_asynccnt 0" ::: "memory");
    }
    __syncthreads();

    // Ls = -A  (negate in place)
    for (int idx = tid; idx < N * N; idx += 1024) Ls[idx] = -Ls[idx];
    __syncthreads();

    // colsum[j] = sum_i A[i][j] = -sum_i Ls[i][j]
    if (tid < N) {
        float s = 0.f;
        for (int i = 0; i < N; ++i) s += Ls[i * N + tid];
        fcol[tid] = -s;
    }
    __syncthreads();
    // L diag = colsum; root row L[0][j] = f_i[b][j]
    if (tid < N) {
        Ls[tid * N + tid] = fcol[tid];
        Ls[tid]           = fi[(size_t)b * N + tid];
    }
    __syncthreads();

    // in-place Gauss-Jordan inversion (no pivoting)
    for (int k = 0; k < N; ++k) {
        if (tid < N) fcol[tid] = Ls[tid * N + k];       // save column k
        if (tid == 0) piv[0] = 1.0f / Ls[k * N + k];
        __syncthreads();
        const float pinv = piv[0];
        if (tid < N) Ls[k * N + tid] = (tid == k) ? pinv : Ls[k * N + tid] * pinv;
        __syncthreads();
        #pragma unroll
        for (int q = 0; q < 8; ++q) {
            int i = wave * 8 + q;
            if (i == k) continue;
            float f = fcol[i];
            #pragma unroll
            for (int c = 0; c < 8; ++c) {
                int j = lane + 32 * c;
                float pv  = Ls[k * N + j];
                float cur = (j == k) ? 0.f : Ls[i * N + j];
                Ls[i * N + j] = cur - f * pv;
            }
        }
        __syncthreads();
    }

    // df[b, j, 0]   = f_i[b,j] * Linv[j,0]
    // df[b, j, i+1] = A[b,i,j] * ((j>0)*Linv[j,j] - (i>0)*Linv[j,i])
    float* out = dfo + (size_t)b * N * (N + 1);
    for (int q = 0; q < 8; ++q) {
        int j = wave * 8 + q;
        float fij = fi[(size_t)b * N + j];
        float ljj = Ls[j * N + j];
        for (int i = lane; i < N + 1; i += 32) {
            float v;
            if (i == 0) {
                v = fij * Ls[j * N];
            } else {
                int ii = i - 1;
                float a  = Ab[(size_t)ii * N + j];
                float t1 = (j  > 0) ? ljj : 0.f;
                float t2 = (ii > 0) ? Ls[j * N + ii] : 0.f;
                v = a * (t1 - t2);
            }
            out[(size_t)j * (N + 1) + i] = v;
        }
    }
}

extern "C" void kernel_launch(void* const* d_in, const int* in_sizes, int n_in,
                              void* d_out, int out_size, void* d_ws, size_t ws_size,
                              hipStream_t stream)
{
    const float* inp = (const float*)d_in[0];
    const float* Wk  = (const float*)d_in[1];
    const float* bk  = (const float*)d_in[2];
    const float* Wq  = (const float*)d_in[3];
    const float* bq  = (const float*)d_in[4];
    const float* Wr  = (const float*)d_in[5];
    const float* br  = (const float*)d_in[6];

    char* ws = (char*)d_ws;
    __bf16* Xb  = (__bf16*)(ws + OFF_XB);
    __bf16* Wkb = (__bf16*)(ws + OFF_WKB);
    __bf16* Wqb = (__bf16*)(ws + OFF_WQB);
    __bf16* Kb  = (__bf16*)(ws + OFF_KB);
    __bf16* Qb  = (__bf16*)(ws + OFF_QB);
    float*  fib = (float*)(ws + OFF_FI);
    float*  Agb = (float*)(ws + OFF_AG);
    float*  dfo = (float*)d_out;

    // fp32 -> bf16 staging
    cvt_bf16_kernel<<<12288, 256, 0, stream>>>(inp, Xb, BT * Dn);
    cvt_bf16_kernel<<<1024, 256, 0, stream>>>(Wk, Wkb, Dn * Dn);
    cvt_bf16_kernel<<<1024, 256, 0, stream>>>(Wq, Wqb, Dn * Dn);

    // root scores (fp32 GEMV)
    fi_kernel<<<1536, 256, 0, stream>>>(inp, Wr, br, fib);

    // key = X Wk^T + bk ; query = (X Wq^T + bq) / sqrt(D)
    gemm_kq_kernel<<<768, 256, 0, stream>>>(Xb, Wkb, bk, 1.0f, Kb);
    gemm_kq_kernel<<<768, 256, 0, stream>>>(Xb, Wqb, bq, 0.03125f, Qb);

    // A = exp(Q K^T) with zero diagonal
    gemm_attn_kernel<<<384, 256, 0, stream>>>(Qb, Kb, Agb);

    // Laplacian, inverse, marginals -> df   (256KB matrix in LDS per WGP)
    const size_t smem_bytes = (size_t)(65536 + 256 + 16) * sizeof(float);
    tree_kernel<<<Bn, 1024, smem_bytes, stream>>>(Agb, fib, dfo);
}
